// FeatureQuantizer_26568667693217
// MI455X (gfx1250) — matrix operand order repaired and checked by reading, not compiled, not run on hardware
//
#include <hip/hip_runtime.h>

typedef __attribute__((ext_vector_type(8)))  __bf16 v8bf;
typedef __attribute__((ext_vector_type(16))) __bf16 v16bf;
typedef __attribute__((ext_vector_type(8)))  float  v8f;

#define DDIM 768
#define TP1 257
#define BATCH 64
#define KC 512
#define KF 4096
#define KTOT 4608
#define NROWS (TP1*BATCH)      /* 16448 total query rows (row n = t*64+b) */
#define NFEAT (256*BATCH)      /* 16384 feature query rows */
#define FMAXV 3.402823466e+38f
#define LDS_STRIDE 40          /* 32 data bf16 + 8 pad -> 80B row stride, bank-conflict-free */

/* d_out layout (fp32 elements): loss | quantized | perplexity | indices | distances */
#define Q_OFF    1
#define Q_SIZE   (NROWS*DDIM)            /* 12,632,064 */
#define PERP_OFF (1 + Q_SIZE)            /* 12,632,065 */
#define ENC_OFF  (PERP_OFF + 1)          /* 12,632,066 */
#define DIST_OFF (ENC_OFF + NROWS)       /* 12,648,514 */

#define PAD_N (BATCH*KF + BATCH*256*KC)  /* 8,650,752 FMAX pad elements */

/* async global -> LDS 16B copy, tracked by ASYNCcnt (CDNA5) */
__device__ __forceinline__ void async_copy_b128(unsigned lds_off, const void* gaddr) {
    asm volatile("global_load_async_to_lds_b128 %0, %1, off"
                 :: "v"(lds_off), "v"(gaddr) : "memory");
}
__device__ __forceinline__ void wait_async0() {
    asm volatile("s_wait_asynccnt 0x0" ::: "memory");
}
__device__ __forceinline__ unsigned lds_off_of(const void* p) {
    /* generic pointer to __shared__ keeps the LDS byte offset in its low 32 bits */
    return (unsigned)(unsigned long long)p;
}

/* ---------------- init: zero loss, perplexity, histogram ---------------- */
__global__ __launch_bounds__(256) void init_kernel(unsigned* hist, float* out) {
    for (int j = threadIdx.x; j < KTOT; j += 256) hist[j] = 0u;
    if (threadIdx.x == 0) { out[0] = 0.0f; out[PERP_OFF] = 0.0f; }
}

/* ---------------- fp32 -> bf16 hi/lo split (Ootomo) ---------------- */
__global__ __launch_bounds__(256) void split_kernel(const float* __restrict__ src,
                                                    __bf16* __restrict__ hi,
                                                    __bf16* __restrict__ lo, int n) {
    int i = blockIdx.x * 256 + threadIdx.x;
    if (i < n) {
        float x = src[i];
        __bf16 h = (__bf16)x;
        hi[i] = h;
        lo[i] = (__bf16)(x - (float)h);
    }
}

/* ---------------- exact fp32 row norms ---------------- */
__global__ __launch_bounds__(256) void row_norm_kernel(const float* __restrict__ src,
                                                       float* __restrict__ out) {
    __shared__ float red[256];
    const float* p = src + (size_t)blockIdx.x * DDIM;
    float s = 0.0f;
    for (int d = threadIdx.x; d < DDIM; d += 256) { float v = p[d]; s += v * v; }
    red[threadIdx.x] = s; __syncthreads();
    for (int off = 128; off > 0; off >>= 1) {
        if (threadIdx.x < off) red[threadIdx.x] += red[threadIdx.x + off];
        __syncthreads();
    }
    if (threadIdx.x == 0) out[blockIdx.x] = red[0];
}

/* ---------------- FMAX padding of cross-codebook distance slots ---------------- */
__global__ __launch_bounds__(256) void pad_kernel(float* __restrict__ dist) {
    int i = blockIdx.x * 256 + threadIdx.x;
    if (i >= PAD_N) return;
    int b, t, j;
    if (i < BATCH * KF) {            /* t==0 rows: pad feature slots [512,4608) */
        b = i >> 12; j = KC + (i & (KF - 1)); t = 0;
    } else {                          /* t>=1 rows: pad class slots [0,512) */
        int r = i - BATCH * KF;
        b = r / (256 * KC);
        int q = r % (256 * KC);
        t = 1 + (q >> 9); j = q & (KC - 1);
    }
    dist[(size_t)(b * TP1 + t) * KTOT + j] = FMAXV;
}

/* ---------------- WMMA distance GEMM: dist = ||x||^2 + ||e||^2 - 2 x.e ----------------
 * 3xBF16 split accumulation: x.e ~= xh.eh + xh.el + xl.eh  (f32 WMMA accumulator).
 * Block tile (32*MT) x 128; 8 waves arranged 2(M) x 4(N), each wave MT x 2 tiles of 16x16.
 * Tiles staged once per block into LDS via async global->LDS B128 copies (ASYNCcnt),
 * double-buffered over K so fills overlap WMMA. */
template<int MT>
__global__ __launch_bounds__(256) void vq_gemm_kernel(
    const __bf16* __restrict__ Xh, const __bf16* __restrict__ Xl,
    const __bf16* __restrict__ Eh, const __bf16* __restrict__ El,
    const float* __restrict__ xnorm, const float* __restrict__ enorm,
    float* __restrict__ dist, int rowStart, int colOff)
{
    constexpr int BM = 32 * MT;
    __shared__ __bf16 sAh[2][BM][LDS_STRIDE];
    __shared__ __bf16 sAl[2][BM][LDS_STRIDE];
    __shared__ __bf16 sBh[2][128][LDS_STRIDE];
    __shared__ __bf16 sBl[2][128][LDS_STRIDE];

    const int wid  = threadIdx.x >> 5;
    const int lane = threadIdx.x & 31;
    const int half = lane >> 4;
    const int l16  = lane & 15;
    const int wOffM = (wid >> 2) * (16 * MT);
    const int wOffN = (wid & 3) * 32;
    const int rowBase = rowStart + blockIdx.y * BM;
    const int colBase = blockIdx.x * 128;

    /* issue async fill of buffer `buf` with K-chunk starting at k0 */
    auto fill = [&](int buf, int k0) {
        const int NCH = BM * 8 + 1024;   /* A: BM rows x 4 quads x {h,l}; B: 128 x 4 x {h,l} */
        for (int c = threadIdx.x; c < NCH; c += 256) {
            const __bf16* src;
            const __bf16* ldst;
            if (c < BM * 8) {
                int row = c >> 3, sub = c & 7, m = sub >> 2, quad = sub & 3;
                src  = (m ? Xl : Xh) + (size_t)(rowBase + row) * DDIM + k0 + quad * 8;
                ldst = m ? &sAl[buf][row][quad * 8] : &sAh[buf][row][quad * 8];
            } else {
                int c2 = c - BM * 8;
                int col = c2 >> 3, sub = c2 & 7, m = sub >> 2, quad = sub & 3;
                src  = (m ? El : Eh) + (size_t)(colBase + col) * DDIM + k0 + quad * 8;
                ldst = m ? &sBl[buf][col][quad * 8] : &sBh[buf][col][quad * 8];
            }
            async_copy_b128(lds_off_of(ldst), src);
        }
    };

    v8f acc[MT][2] = {};

    fill(0, 0);
    wait_async0();
    __syncthreads();

    for (int kt = 0; kt < DDIM / 32; ++kt) {
        const int cur = kt & 1;
        if (kt + 1 < DDIM / 32) fill(cur ^ 1, (kt + 1) * 32);

        v16bf bh[2], bl[2];
#pragma unroll
        for (int nt = 0; nt < 2; ++nt) {
            int bc = wOffN + nt * 16 + l16;            /* B col; half selects K-subchunks */
            v8bf b0 = *(const v8bf*)&sBh[cur][bc][half * 8];
            v8bf b1 = *(const v8bf*)&sBh[cur][bc][16 + half * 8];
            bh[nt] = __builtin_shufflevector(b0, b1, 0,1,2,3,4,5,6,7,8,9,10,11,12,13,14,15);
            v8bf c0 = *(const v8bf*)&sBl[cur][bc][half * 8];
            v8bf c1 = *(const v8bf*)&sBl[cur][bc][16 + half * 8];
            bl[nt] = __builtin_shufflevector(c0, c1, 0,1,2,3,4,5,6,7,8,9,10,11,12,13,14,15);
        }
#pragma unroll
        for (int mt = 0; mt < MT; ++mt) {
            int ar = wOffM + mt * 16 + l16;            /* A row; half selects K-subchunks */
            v8bf ah0 = *(const v8bf*)&sAh[cur][ar][half * 8];
            v8bf ah1 = *(const v8bf*)&sAh[cur][ar][16 + half * 8];
            v8bf al0 = *(const v8bf*)&sAl[cur][ar][half * 8];
            v8bf al1 = *(const v8bf*)&sAl[cur][ar][16 + half * 8];
            v16bf ah = __builtin_shufflevector(ah0, ah1, 0,1,2,3,4,5,6,7,8,9,10,11,12,13,14,15);
            v16bf al = __builtin_shufflevector(al0, al1, 0,1,2,3,4,5,6,7,8,9,10,11,12,13,14,15);
#pragma unroll
            for (int nt = 0; nt < 2; ++nt) {
                acc[mt][nt] = __builtin_amdgcn_wmma_f32_16x16x32_bf16(
                    false, ah, false, bh[nt], (short)0, acc[mt][nt], false, false);
                acc[mt][nt] = __builtin_amdgcn_wmma_f32_16x16x32_bf16(
                    false, ah, false, bl[nt], (short)0, acc[mt][nt], false, false);
                acc[mt][nt] = __builtin_amdgcn_wmma_f32_16x16x32_bf16(
                    false, al, false, bh[nt], (short)0, acc[mt][nt], false, false);
            }
        }

        wait_async0();     /* next buffer filled (this wave's share) */
        __syncthreads();   /* all waves done filling + done reading `cur` */
    }

    /* epilogue: C element i of lane -> row = i + 8*half, col = lane%16 */
#pragma unroll
    for (int mt = 0; mt < MT; ++mt) {
#pragma unroll
        for (int nt = 0; nt < 2; ++nt) {
            int col = colBase + wOffN + nt * 16 + l16;
            float en = enorm[colOff + col];
#pragma unroll
            for (int i = 0; i < 8; ++i) {
                int ng = rowBase + wOffM + mt * 16 + i + 8 * half;  /* global query row */
                int b = ng & 63, t = ng >> 6;
                dist[(size_t)(b * TP1 + t) * KTOT + colOff + col] =
                    xnorm[ng] + en - 2.0f * acc[mt][nt][i];
            }
        }
    }
}

/* ---------------- per-row argmin over padded 4608-wide distance row ---------------- */
__global__ __launch_bounds__(256) void argmin_kernel(const float* __restrict__ dist,
                                                     int* __restrict__ idxw,
                                                     float* __restrict__ minv,
                                                     float* __restrict__ out,
                                                     unsigned* __restrict__ hist) {
    __shared__ float sv[256];
    __shared__ int   si[256];
    int n = blockIdx.x;
    int b = n & 63, t = n >> 6;
    const float* p = dist + (size_t)(b * TP1 + t) * KTOT;
    float bv = FMAXV; int bi = KTOT;
    for (int j = threadIdx.x; j < KTOT; j += 256) {
        float v = p[j];
        if (v < bv) { bv = v; bi = j; }
    }
    sv[threadIdx.x] = bv; si[threadIdx.x] = bi; __syncthreads();
    for (int off = 128; off > 0; off >>= 1) {
        if (threadIdx.x < off) {
            float ov = sv[threadIdx.x + off]; int oi = si[threadIdx.x + off];
            if (ov < sv[threadIdx.x] || (ov == sv[threadIdx.x] && oi < si[threadIdx.x])) {
                sv[threadIdx.x] = ov; si[threadIdx.x] = oi;
            }
        }
        __syncthreads();
    }
    if (threadIdx.x == 0) {
        idxw[n] = si[0];
        minv[n] = sv[0];
        out[ENC_OFF + n] = (float)si[0];      /* global index: class in [0,512), feat+512 */
        atomicAdd(&hist[si[0]], 1u);
    }
}

/* ---------------- deterministic loss reduction ---------------- */
__global__ __launch_bounds__(256) void loss_kernel(const float* __restrict__ minv,
                                                   float* __restrict__ out) {
    __shared__ float red[256];
    const float sc_c = 0.25f / (64.0f * 768.0f);
    const float sc_f = 0.25f / (16384.0f * 768.0f);
    float s = 0.0f;
    for (int n = threadIdx.x; n < NROWS; n += 256)
        s += minv[n] * (n < BATCH ? sc_c : sc_f);
    red[threadIdx.x] = s; __syncthreads();
    for (int off = 128; off > 0; off >>= 1) {
        if (threadIdx.x < off) red[threadIdx.x] += red[threadIdx.x + off];
        __syncthreads();
    }
    if (threadIdx.x == 0) out[0] = red[0];
}

/* ---------------- perplexity (block 0: class, block 1: feature) ---------------- */
__global__ __launch_bounds__(256) void perp_kernel(const unsigned* __restrict__ hist,
                                                   float* __restrict__ out) {
    __shared__ float red[256];
    int K   = blockIdx.x ? KF : KC;
    int off = blockIdx.x ? KC : 0;
    float invN = blockIdx.x ? (1.0f / 16384.0f) : (1.0f / 64.0f);
    float h = 0.0f;
    for (int k = threadIdx.x; k < K; k += 256) {
        float pv = (float)hist[off + k] * invN;
        h += pv * logf(pv + 1e-10f);
    }
    red[threadIdx.x] = h; __syncthreads();
    for (int o = 128; o > 0; o >>= 1) {
        if (threadIdx.x < o) red[threadIdx.x] += red[threadIdx.x + o];
        __syncthreads();
    }
    if (threadIdx.x == 0) atomicAdd(&out[PERP_OFF], expf(-red[0]));
}

/* ---------------- gather quantized features (exact fp32 codebook rows) ---------------- */
__global__ __launch_bounds__(256) void gather_kernel(const int* __restrict__ idxw,
                                                     const float* __restrict__ ccb,
                                                     const float* __restrict__ fcb,
                                                     float* __restrict__ out) {
    int n = blockIdx.x;
    int id = idxw[n];
    const float* src = (id < KC) ? (ccb + (size_t)id * DDIM)
                                 : (fcb + (size_t)(id - KC) * DDIM);
    float* dst = out + Q_OFF + (size_t)n * DDIM;
    for (int d = threadIdx.x; d < DDIM; d += 256) dst[d] = src[d];
}

extern "C" void kernel_launch(void* const* d_in, const int* in_sizes, int n_in,
                              void* d_out, int out_size, void* d_ws, size_t ws_size,
                              hipStream_t stream) {
    const float* feats = (const float*)d_in[0];   /* [257,64,768] */
    const float* ccb   = (const float*)d_in[1];   /* [512,768]   */
    const float* fcb   = (const float*)d_in[2];   /* [4096,768]  */
    float* out = (float*)d_out;
    char* ws = (char*)d_ws;

    /* workspace layout (bytes) */
    __bf16*   Xh    = (__bf16*)(ws + 0);          /* 16448*768 bf16 */
    __bf16*   Xl    = (__bf16*)(ws + 25264128);
    __bf16*   Eh    = (__bf16*)(ws + 50528256);   /* 4608*768 bf16 (class rows 0..511) */
    __bf16*   El    = (__bf16*)(ws + 57606144);
    float*    xnorm = (float*)(ws + 64684032);    /* 16448 f32 */
    float*    enorm = (float*)(ws + 64749824);    /* 4608 f32  */
    unsigned* hist  = (unsigned*)(ws + 64768256); /* 4608 u32  */
    float*    minv  = (float*)(ws + 64786688);    /* 16448 f32 */
    int*      idxw  = (int*)(ws + 64852480);      /* 16448 i32 */

    float* dist = out + DIST_OFF;

    init_kernel<<<1, 256, 0, stream>>>(hist, out);

    split_kernel<<<(NROWS * DDIM + 255) / 256, 256, 0, stream>>>(feats, Xh, Xl, NROWS * DDIM);
    split_kernel<<<(KC * DDIM + 255) / 256, 256, 0, stream>>>(ccb, Eh, El, KC * DDIM);
    split_kernel<<<(KF * DDIM + 255) / 256, 256, 0, stream>>>(
        fcb, Eh + (size_t)KC * DDIM, El + (size_t)KC * DDIM, KF * DDIM);

    row_norm_kernel<<<NROWS, 256, 0, stream>>>(feats, xnorm);
    row_norm_kernel<<<KC, 256, 0, stream>>>(ccb, enorm);
    row_norm_kernel<<<KF, 256, 0, stream>>>(fcb, enorm + KC);

    pad_kernel<<<(PAD_N + 255) / 256, 256, 0, stream>>>(dist);

    /* class: 64x512 (MT=2 -> 64x128 block tiles) */
    vq_gemm_kernel<2><<<dim3(KC / 128, 1), 256, 0, stream>>>(
        Xh, Xl, Eh, El, xnorm, enorm, dist, 0, 0);
    /* feature: 16384x4096 (MT=4 -> 128x128 block tiles) */
    vq_gemm_kernel<4><<<dim3(KF / 128, NFEAT / 128), 256, 0, stream>>>(
        Xh, Xl, Eh + (size_t)KC * DDIM, El + (size_t)KC * DDIM,
        xnorm, enorm, dist, BATCH, KC);

    argmin_kernel<<<NROWS, 256, 0, stream>>>(dist, idxw, minv, out, hist);
    loss_kernel<<<1, 256, 0, stream>>>(minv, out);
    perp_kernel<<<2, 256, 0, stream>>>(hist, out);
    gather_kernel<<<NROWS, 256, 0, stream>>>(idxw, ccb, fcb, out);
}
